// Grapher_42580305773081
// MI455X (gfx1250) — matrix operand order repaired and checked by reading, not compiled, not run on hardware
//
#include <hip/hip_runtime.h>

// ---------------------------------------------------------------------------
// CDNA5 (gfx1250) Grapher block: fc1 -> kNN(graph) -> MRConv -> fc2, all GEMMs
// via v_wmma_f32_16x16x32_bf16 (wave32, fp32 accumulate). kNN key fragments
// are staged global->LDS with double-buffered global_load_async_to_lds_b128
// (ASYNCcnt) shared by all 4 waves of the workgroup.
// Shapes: B=8, C=192, HID=384, N=56*56=3136, k=9.
// ---------------------------------------------------------------------------

typedef __attribute__((ext_vector_type(16))) __bf16 v16bf;
typedef __attribute__((ext_vector_type(8)))  float  v8f;

union Frag {
  v16bf v;
  uint4 q[2];
  unsigned int u[8];
};

#define B_DIM   8
#define C_DIM   192
#define HID_DIM 384
#define N_PTS   3136
#define NBLK    196      // N_PTS / 16
#define KCH_C   6        // C_DIM / 32 (K-chunks for the distance GEMM)
#define KNN     9
#define FRAG_B  1024     // bytes per fragment (32 lanes x 32B)

// float -> bf16, round-to-nearest-even
__device__ inline unsigned int f2bf(float f) {
  unsigned int u = __float_as_uint(f);
  return (u + 0x7FFFu + ((u >> 16) & 1u)) >> 16;
}
__device__ inline unsigned int pack2(float lo, float hi) {
  return f2bf(lo) | (f2bf(hi) << 16);
}
__device__ inline float gelu_exact(float x) {
  return 0.5f * x * (1.0f + erff(x * 0.70710678118654752f));
}

// ---------------------------------------------------------------------------
// Generic fp32-in / fp32-out GEMM  Y[b] = W(MxK) @ X[b](KxN), bf16 WMMA compute.
// Block: 128 threads (4 waves). Tile: 64(M) x 64(N). Wave w owns M-rows
// [mb+16w, mb+16w+16) and all four 16-wide N tiles -> 4 accumulators.
// Epilogue modes:
//   0: y = BN0(y + bias)                                  (fc1 -> h)
//   1: y = gelu(BN1(gelu(BN0(y + bias))))                 (Wg  -> g)
//   2: y = BN0(y + bias) + resid                          (fc2 -> out)
// ---------------------------------------------------------------------------
__global__ __launch_bounds__(128)
void gemm_bn_kernel(const float* __restrict__ Wmat,   // [M][K]
                    const float* __restrict__ X,      // [B][K][N]
                    float* __restrict__ Y,            // [B][M][N]
                    const float* __restrict__ bias,   // [M]
                    const float* __restrict__ sg0, const float* __restrict__ sb0,
                    const float* __restrict__ sm0, const float* __restrict__ sv0,
                    const float* __restrict__ sg1, const float* __restrict__ sb1,
                    const float* __restrict__ sm1, const float* __restrict__ sv1,
                    const float* __restrict__ resid,  // [B][M][N] or null
                    int M, int K, int N, int mode)
{
  const int b    = blockIdx.z;
  const int mb   = blockIdx.y * 64;
  const int nb   = blockIdx.x * 64;
  const int tid  = threadIdx.x;
  const int lane = tid & 31;
  const int wave = tid >> 5;

  __shared__ __align__(16) unsigned short As[64 * 32];  // [m][k], k contiguous
  __shared__ __align__(16) unsigned short Bs[64 * 32];  // [n][k], k contiguous

  v8f acc[4];
  const v8f vzero = {0.f, 0.f, 0.f, 0.f, 0.f, 0.f, 0.f, 0.f};
#pragma unroll
  for (int i = 0; i < 4; ++i) acc[i] = vzero;

  for (int kc = 0; kc < K; kc += 32) {
    // ---- stage A tile: 64(M) x 32(K), fp32 -> bf16 -------------------------
    {
      const int m    = tid & 63;
      const int koff = (tid >> 6) * 16;               // 0 or 16
      const float* src = Wmat + (size_t)(mb + m) * K + kc + koff;
      unsigned int t[8];
#pragma unroll
      for (int i = 0; i < 8; ++i) t[i] = pack2(src[2 * i], src[2 * i + 1]);
      uint4* dst = (uint4*)&As[m * 32 + koff];
      dst[0] = make_uint4(t[0], t[1], t[2], t[3]);
      dst[1] = make_uint4(t[4], t[5], t[6], t[7]);
      // prefetch next K-chunk of W into L2/L0 (global_prefetch_b8)
      if (kc + 32 < K) __builtin_prefetch(src + 32, 0, 3);
    }
    // ---- stage B tile: 32(K) x 64(N) -> LDS as [n][k] ----------------------
    {
      const int n   = tid & 63;
      const int kp0 = (tid >> 6) * 8;                 // k-pair base: 0 or 8
      const float* src = X + ((size_t)b * K + kc) * N + nb + n;
      unsigned int t[8];
#pragma unroll
      for (int i = 0; i < 8; ++i) {
        const int k = (kp0 + i) * 2;
        t[i] = pack2(src[(size_t)k * N], src[(size_t)(k + 1) * N]);
      }
      uint4* dst = (uint4*)&Bs[n * 32 + kp0 * 2];
      dst[0] = make_uint4(t[0], t[1], t[2], t[3]);
      dst[1] = make_uint4(t[4], t[5], t[6], t[7]);
      // prefetch next K-chunk of X
      if (kc + 32 < K) __builtin_prefetch(src + (size_t)32 * N, 0, 3);
    }
    __syncthreads();

    // ---- A fragment for this wave (ISA 16-bit 16x32 layout) ----------------
    Frag af;
    {
      const int m     = lane & 15;
      const int koffA = (lane < 16) ? 0 : 8;          // K 0-7/16-23 vs 8-15/24-31
      const unsigned short* row = &As[(wave * 16 + m) * 32];
      af.q[0] = *(const uint4*)&row[koffA];
      af.q[1] = *(const uint4*)&row[koffA + 16];
    }
    // ---- 4 WMMAs over the four N tiles -------------------------------------
#pragma unroll
    for (int nt = 0; nt < 4; ++nt) {
      Frag bf;
      const int n     = nt * 16 + (lane & 15);
      const int koffB = (lane < 16) ? 0 : 8;          // K 0-7/16-23 vs 8-15/24-31
      const uint4* p  = (const uint4*)&Bs[n * 32 + koffB];
      bf.q[0] = p[0];
      bf.q[1] = p[2];
      acc[nt] = __builtin_amdgcn_wmma_f32_16x16x32_bf16(
          false, af.v, false, bf.v, (short)0, acc[nt], false, false);
    }
    __syncthreads();
  }

  // ---- fused epilogue ------------------------------------------------------
  const int mrow0 = mb + wave * 16 + ((lane >> 4) << 3);   // + v gives row
  const int ncol0 = nb + (lane & 15);
#pragma unroll
  for (int nt = 0; nt < 4; ++nt) {
    const int n = ncol0 + nt * 16;
#pragma unroll
    for (int v = 0; v < 8; ++v) {
      const int m = mrow0 + v;
      float y = acc[nt][v] + bias[m];
      const float s0 = sg0[m] * rsqrtf(sv0[m] + 1e-5f);
      y = (y - sm0[m]) * s0 + sb0[m];
      if (mode == 1) {
        y = gelu_exact(y);
        const float s1 = sg1[m] * rsqrtf(sv1[m] + 1e-5f);
        y = (y - sm1[m]) * s1 + sb1[m];
        y = gelu_exact(y);
      } else if (mode == 2) {
        y += resid[((size_t)b * M + m) * N + n];
      }
      Y[((size_t)b * M + m) * N + n] = y;
    }
  }
}

// ---------------------------------------------------------------------------
// Normalize h along C per point; emit pre-swizzled bf16 fragments of hn:
//  hnA: A-matrix (16 pts x 32 ch) fragments, lane-order storage (32B/lane)
//  hnB: B-matrix (32 ch x 16 pts) fragments, lane-order storage
//  sq : sum(hn^2) per point (== 1 unless norm ~ 0)
// One block per (point-block p, batch b). Block = 128 threads.
// ---------------------------------------------------------------------------
__global__ __launch_bounds__(128)
void normalize_swizzle_kernel(const float* __restrict__ h,
                              uint4* __restrict__ hnA,
                              uint4* __restrict__ hnB,
                              float* __restrict__ sq)
{
  const int p   = blockIdx.x;          // point block (16 points)
  const int b   = blockIdx.y;
  const int tid = threadIdx.x;

  __shared__ float tile[C_DIM * 16];   // [c][j]
  __shared__ float rn[16];

  const float* hb = h + (size_t)b * C_DIM * N_PTS;
  for (int e = tid; e < C_DIM * 16; e += 128) {
    const int c = e >> 4, j = e & 15;
    tile[e] = hb[(size_t)c * N_PTS + p * 16 + j];
  }
  __syncthreads();

  if (tid < 16) {
    float s = 0.f;
    for (int c = 0; c < C_DIM; ++c) {
      const float t = tile[c * 16 + tid];
      s += t * t;
    }
    const float nrm = fmaxf(sqrtf(s), 1e-12f);
    const float r = 1.0f / nrm;
    rn[tid] = r;
    sq[(size_t)b * N_PTS + p * 16 + tid] = s * r * r;
  }
  __syncthreads();

  const size_t fragbase = ((size_t)b * NBLK + p) * KCH_C;   // in 1KB fragments
  for (int s = tid; s < KCH_C * 32; s += 128) {
    const int qf = s >> 5;             // K-chunk
    const int L  = s & 31;             // lane slot
    const int j  = L & 15;
    const float r = rn[j];
    unsigned int u[8];
    // ---- A-swizzled: lane<16 -> K 0-7,16-23 ; lane>=16 -> K 8-15,24-31 -----
    {
      const int k0 = qf * 32 + ((L < 16) ? 0 : 8);
      float vals[16];
#pragma unroll
      for (int i = 0; i < 8; ++i) vals[i]     = tile[(k0 + i) * 16 + j] * r;
#pragma unroll
      for (int i = 0; i < 8; ++i) vals[8 + i] = tile[(k0 + 16 + i) * 16 + j] * r;
#pragma unroll
      for (int t = 0; t < 8; ++t) u[t] = pack2(vals[2 * t], vals[2 * t + 1]);
      uint4* dst = hnA + (fragbase + qf) * 64 + L * 2;
      dst[0] = make_uint4(u[0], u[1], u[2], u[3]);
      dst[1] = make_uint4(u[4], u[5], u[6], u[7]);
    }
    // ---- B-swizzled: lane<16 -> K 0-7,16-23 ; lane>=16 -> K 8-15,24-31 -----
    {
      const int k0 = qf * 32 + ((L < 16) ? 0 : 8);
      float vals[16];
#pragma unroll
      for (int i = 0; i < 16; ++i) vals[i] = tile[(k0 + i + ((i >> 3) << 3)) * 16 + j] * r;
#pragma unroll
      for (int t = 0; t < 8; ++t) u[t] = pack2(vals[2 * t], vals[2 * t + 1]);
      uint4* dst = hnB + (fragbase + qf) * 64 + L * 2;
      dst[0] = make_uint4(u[0], u[1], u[2], u[3]);
      dst[1] = make_uint4(u[4], u[5], u[6], u[7]);
    }
  }
}

// ---------------------------------------------------------------------------
// Fused Gram-matrix + top-k. Per query block (16 pts) held in A fragments,
// stream all 196 key blocks: key fragments (6KB/block, bf16, pre-swizzled)
// are copied global->LDS with global_load_async_to_lds_b128, double-buffered
// (ASYNCcnt / s_wait_asynccnt) and shared by all 4 waves; 6 WMMAs per key
// block; score = 2*inner - sq_j; per-query top-9 kept in registers.
// ---------------------------------------------------------------------------
__global__ __launch_bounds__(128)
void knn_topk_kernel(const uint4* __restrict__ hnA,
                     const char*  __restrict__ hnB,   // byte base (async copy)
                     const float* __restrict__ sq,
                     int* __restrict__ nbr)
{
  const int tid  = threadIdx.x;
  const int w    = tid >> 5;
  const int lane = tid & 31;
  const int b    = blockIdx.y;
  const int qblk = blockIdx.x * 4 + w;

  __shared__ __align__(16) char Bsh[2][KCH_C * FRAG_B];  // 2 x 6KB key frags
  __shared__ float scores[4][16][16];
  __shared__ float sqs[4][16];

  // LDS byte offsets for async-copy destinations (low 32 bits of generic addr)
  const unsigned ldsB[2] = { (unsigned)(size_t)(&Bsh[0][0]),
                             (unsigned)(size_t)(&Bsh[1][0]) };
  const unsigned long long gbase = (unsigned long long)(size_t)hnB;

  // each thread copies 3 x 16B; per wave: 3 async instructions (1536B), 4 waves
  // cover the 6KB key block.
#define ISSUE_KEY_COPY(JB, BUF)                                               \
  do {                                                                        \
    unsigned goff = (unsigned)(((b * NBLK) + (JB)) * (KCH_C * FRAG_B)) +      \
                    (unsigned)tid * 16u;                                      \
    unsigned loff = ldsB[(BUF)] + (unsigned)tid * 16u;                        \
    asm volatile("global_load_async_to_lds_b128 %0, %1, %2"                   \
                 :: "v"(loff), "v"(goff), "s"(gbase) : "memory");             \
    asm volatile("global_load_async_to_lds_b128 %0, %1, %2"                   \
                 :: "v"(loff + 2048u), "v"(goff + 2048u), "s"(gbase)          \
                 : "memory");                                                 \
    asm volatile("global_load_async_to_lds_b128 %0, %1, %2"                   \
                 :: "v"(loff + 4096u), "v"(goff + 4096u), "s"(gbase)          \
                 : "memory");                                                 \
  } while (0)

  // query A-fragments (held in registers for the whole sweep)
  Frag af[KCH_C];
  {
    const uint4* pa = hnA + ((size_t)b * NBLK + qblk) * KCH_C * 64;
#pragma unroll
    for (int q = 0; q < KCH_C; ++q) {
      af[q].q[0] = pa[q * 64 + lane * 2];
      af[q].q[1] = pa[q * 64 + lane * 2 + 1];
    }
  }

  float bestv[KNN];
  int   besti[KNN];
#pragma unroll
  for (int t = 0; t < KNN; ++t) { bestv[t] = -3.4e38f; besti[t] = 0; }

  const v8f vzero = {0.f, 0.f, 0.f, 0.f, 0.f, 0.f, 0.f, 0.f};
  const int mloc = (lane >> 4) << 3;
  const int ncol = lane & 15;

  ISSUE_KEY_COPY(0, 0);                       // prologue: key block 0 -> buf 0

  for (int jb = 0; jb < NBLK; ++jb) {
    if (jb + 1 < NBLK) {
      ISSUE_KEY_COPY(jb + 1, (jb + 1) & 1);   // prefetch next key block
      // async loads complete in order: <=3 outstanding => jb's copy landed
      asm volatile("s_wait_asynccnt 0x3" ::: "memory");
    } else {
      asm volatile("s_wait_asynccnt 0x0" ::: "memory");
    }
    __syncthreads();                          // key block jb visible to all waves

    const char* bufp = &Bsh[jb & 1][0];
    v8f c = vzero;
#pragma unroll
    for (int q = 0; q < KCH_C; ++q) {
      Frag bf;
      const uint4* p = (const uint4*)(bufp + q * FRAG_B + lane * 32);
      bf.q[0] = p[0];
      bf.q[1] = p[1];
      c = __builtin_amdgcn_wmma_f32_16x16x32_bf16(
          false, af[q].v, false, bf.v, (short)0, c, false, false);
    }
    if (lane < 16) sqs[w][lane] = sq[(size_t)b * N_PTS + jb * 16 + lane];
#pragma unroll
    for (int v = 0; v < 8; ++v) scores[w][mloc + v][ncol] = c[v];
    // single-wave LDS RAW on scores/sqs: DS ops are in-order per wave
    if (lane < 16) {
      for (int col = 0; col < 16; ++col) {
        const float val = 2.0f * scores[w][lane][col] - sqs[w][col];
        if (val > bestv[KNN - 1]) {
          int ppos = KNN - 1;
          while (ppos > 0 && bestv[ppos - 1] < val) {
            bestv[ppos] = bestv[ppos - 1];
            besti[ppos] = besti[ppos - 1];
            --ppos;
          }
          bestv[ppos] = val;
          besti[ppos] = jb * 16 + col;
        }
      }
    }
    __syncthreads();                          // all waves done with buf[jb&1]
  }
#undef ISSUE_KEY_COPY

  if (lane < 16) {
    const int n = qblk * 16 + lane;
    int* op = nbr + ((size_t)b * N_PTS + n) * KNN;
#pragma unroll
    for (int t = 0; t < KNN; ++t) op[t] = besti[t];
  }
}

// ---------------------------------------------------------------------------
// MRConv gather: cat[b][2c][n] = h[b][c][n]; cat[b][2c+1][n] = max_j h[c][idx_j] - h[c][n]
// One block per (point n, batch b); 192 threads = channels (coalesced gathers).
// ---------------------------------------------------------------------------
__global__ __launch_bounds__(192)
void mrconv_gather_kernel(const float* __restrict__ h,
                          const int* __restrict__ nbr,
                          float* __restrict__ cat)
{
  const int n = blockIdx.x;
  const int b = blockIdx.y;
  const int c = threadIdx.x;

  __shared__ int nb_s[KNN];
  if (c < KNN) nb_s[c] = nbr[((size_t)b * N_PTS + n) * KNN + c];
  __syncthreads();

  const float* hb = h + (size_t)b * C_DIM * N_PTS;
  const float hv = hb[(size_t)c * N_PTS + n];
  float mx = -3.4e38f;
#pragma unroll
  for (int j = 0; j < KNN; ++j)
    mx = fmaxf(mx, hb[(size_t)c * N_PTS + nb_s[j]]);

  float* cb = cat + (size_t)b * (2 * C_DIM) * N_PTS;
  cb[(size_t)(2 * c) * N_PTS + n]     = hv;
  cb[(size_t)(2 * c + 1) * N_PTS + n] = mx - hv;
}

// ---------------------------------------------------------------------------
extern "C" void kernel_launch(void* const* d_in, const int* in_sizes, int n_in,
                              void* d_out, int out_size, void* d_ws, size_t ws_size,
                              hipStream_t stream)
{
  const float* x   = (const float*)d_in[0];
  const float* W1  = (const float*)d_in[1];
  const float* b1  = (const float*)d_in[2];
  const float* g1  = (const float*)d_in[3];
  const float* be1 = (const float*)d_in[4];
  const float* m1  = (const float*)d_in[5];
  const float* v1  = (const float*)d_in[6];
  const float* Wg  = (const float*)d_in[7];
  const float* bg  = (const float*)d_in[8];
  const float* gg  = (const float*)d_in[9];
  const float* beg = (const float*)d_in[10];
  const float* mg  = (const float*)d_in[11];
  const float* vg  = (const float*)d_in[12];
  const float* g2  = (const float*)d_in[13];
  const float* be2 = (const float*)d_in[14];
  const float* m2  = (const float*)d_in[15];
  const float* v2  = (const float*)d_in[16];
  const float* W2  = (const float*)d_in[17];
  const float* b2  = (const float*)d_in[18];
  const float* g3  = (const float*)d_in[19];
  const float* be3 = (const float*)d_in[20];
  const float* m3  = (const float*)d_in[21];
  const float* v3  = (const float*)d_in[22];
  (void)in_sizes; (void)n_in; (void)out_size; (void)ws_size; // k fixed at 9

  char* ws = (char*)d_ws;
  float* h    = (float*)(ws + 0);            // B*C*N f32        = 19,267,584 B
  float* cat  = (float*)(ws + 19267584);     // B*2C*N f32       = 38,535,168 B
  float* g    = (float*)(ws + 57802752);     // B*HID*N f32      = 38,535,168 B
  uint4* hnA  = (uint4*)(ws + 96337920);     // B*196*6*1024 B   =  9,633,792 B
  char*  hnB  = (char*) (ws + 105971712);    //                  =  9,633,792 B
  float* sq   = (float*)(ws + 115605504);    // B*N f32          =    100,352 B
  int*   nbr  = (int*)  (ws + 115705856);    // B*N*9 i32        =    903,168 B

  const dim3 blk128(128);

  // fc1: h = BN1(W1 @ x + b1)
  gemm_bn_kernel<<<dim3(49, 3, 8), blk128, 0, stream>>>(
      W1, x, h, b1, g1, be1, m1, v1,
      nullptr, nullptr, nullptr, nullptr, nullptr,
      C_DIM, C_DIM, N_PTS, 0);

  // normalize + fragment-swizzle hn (bf16), sq
  normalize_swizzle_kernel<<<dim3(NBLK, B_DIM), blk128, 0, stream>>>(
      h, hnA, (uint4*)hnB, sq);

  // fused Gram GEMM + top-9 neighbor indices (async-LDS staged keys)
  knn_topk_kernel<<<dim3(49, B_DIM), blk128, 0, stream>>>(hnA, hnB, sq, nbr);

  // MRConv gather/max -> interleaved cat (B, 2C, N)
  mrconv_gather_kernel<<<dim3(N_PTS, B_DIM), dim3(192), 0, stream>>>(h, nbr, cat);

  // Wg: g = gelu(BN2(gelu(BNg(Wg @ cat + bg))))
  gemm_bn_kernel<<<dim3(49, 6, 8), blk128, 0, stream>>>(
      Wg, cat, g, bg, gg, beg, mg, vg,
      g2, be2, m2, v2, nullptr,
      HID_DIM, 2 * C_DIM, N_PTS, 1);

  // fc2: out = BN3(W2 @ g + b2) + x
  gemm_bn_kernel<<<dim3(49, 3, 8), blk128, 0, stream>>>(
      W2, g, (float*)d_out, b2, g3, be3, m3, v3,
      nullptr, nullptr, nullptr, nullptr, x,
      C_DIM, HID_DIM, N_PTS, 2);
}
